// NemotronHMoE_83236466196644
// MI455X (gfx1250) — matrix operand order repaired and checked by reading, not compiled, not run on hardware
//
#include <hip/hip_runtime.h>
#include <hip/hip_bf16.h>
#include <cstdint>

#define T_TOK 256
#define HDIM  2048
#define NEXP  32
#define IDIM  1024
#define SDIM  8192
#define TOPK  8
#define NGROUP 8
#define GSIZE  (NEXP / NGROUP)   // 4 experts per group
#define TOPKG  4
#define ROUTED_SCALE 2.5f

typedef __attribute__((ext_vector_type(16))) __bf16 v16bf;
typedef __attribute__((ext_vector_type(8)))  float  v8f;

// ---------------- WMMA helpers (CDNA5 16x16x32 bf16, f32 accumulate) ----------------

static __device__ __forceinline__ v8f wmma_bf16(v16bf a, v16bf b, v8f c) {
  // (neg_a, A, neg_b, B, c_mod, C, reuse_a, reuse_b)
  return __builtin_amdgcn_wmma_f32_16x16x32_bf16(false, a, false, b, (short)0, c, false, false);
}

// A-matrix 16x32 bf16 layout: lane holds row (lane&15); K chunks at (lane>>4)*8 and +16.
static __device__ __forceinline__ v16bf load_a_f32row(const float* __restrict__ row, int k0, int lane) {
  const int kb = k0 + ((lane >> 4) << 3);
  v16bf a;
#pragma unroll
  for (int j = 0; j < 8; ++j) a[j] = (__bf16)row[kb + j];
#pragma unroll
  for (int j = 0; j < 8; ++j) a[8 + j] = (__bf16)row[kb + 16 + j];
  return a;
}

static __device__ __forceinline__ v16bf load_a_bf16row(const __bf16* __restrict__ row, int k0, int lane) {
  const int kb = k0 + ((lane >> 4) << 3);
  v16bf a;
#pragma unroll
  for (int j = 0; j < 8; ++j) a[j] = row[kb + j];
#pragma unroll
  for (int j = 0; j < 8; ++j) a[8 + j] = row[kb + 16 + j];
  return a;
}

// B-matrix 32x16 bf16 layout: lane holds column (lane&15); K chunks at (lane>>4)*8 and +16.
// W is row-major [K x ldw] fp32; converted to bf16 in-register.
static __device__ __forceinline__ v16bf load_b_f32(const float* __restrict__ W, int k0, int col,
                                                   int ldw, int lane) {
  const int c  = col + (lane & 15);
  const int kb = k0 + ((lane >> 4) << 3);
  const float* p = W + (size_t)kb * (size_t)ldw + c;
  v16bf b;
#pragma unroll
  for (int j = 0; j < 8; ++j) b[j] = (__bf16)p[(size_t)j * (size_t)ldw];
#pragma unroll
  for (int j = 0; j < 8; ++j) b[8 + j] = (__bf16)p[(size_t)(16 + j) * (size_t)ldw];
  return b;
}

// ---------------- Kernel 1: router (logits -> sigmoid -> biased grouped top-k) ----------------

__global__ void router_kernel(const float* __restrict__ x, const float* __restrict__ gw,
                              const float* __restrict__ bias, float* __restrict__ combine) {
  const int t = blockIdx.x;
  const int e = threadIdx.x;                 // 32 threads = 1 wave, one expert per lane
  const float* xr = x + (size_t)t * HDIM;
  const float* wr = gw + (size_t)e * HDIM;
  float acc = 0.f;
#pragma unroll 4
  for (int h = 0; h < HDIM; ++h) acc += xr[h] * wr[h];
  const float sc = 1.f / (1.f + expf(-acc));

  __shared__ float s_sc[NEXP];
  __shared__ float s_b[NEXP];
  s_sc[e] = sc;
  s_b[e]  = sc + bias[e];
  combine[(size_t)t * NEXP + e] = 0.f;
  __syncthreads();

  if (e == 0) {
    // group score = sum of top-2 biased scores within each group of 4
    float gsc[NGROUP];
    for (int g = 0; g < NGROUP; ++g) {
      float m1 = -1e30f, m2 = -1e30f;
      for (int i = 0; i < GSIZE; ++i) {
        float v = s_b[g * GSIZE + i];
        if (v > m1) { m2 = m1; m1 = v; }
        else if (v > m2) { m2 = v; }
      }
      gsc[g] = m1 + m2;
    }
    bool gsel[NGROUP] = {};
    for (int it = 0; it < TOPKG; ++it) {
      int best = 0; float bv = -1e30f;
      for (int g = 0; g < NGROUP; ++g)
        if (!gsel[g] && gsc[g] > bv) { bv = gsc[g]; best = g; }
      gsel[best] = true;
    }
    float tmp[NEXP];
    for (int i = 0; i < NEXP; ++i) tmp[i] = gsel[i / GSIZE] ? s_b[i] : 0.0f;
    int ids[TOPK]; float wv[TOPK]; float wsum = 0.f;
    for (int it = 0; it < TOPK; ++it) {
      int best = 0; float bv = -1e30f;
      for (int i = 0; i < NEXP; ++i)
        if (tmp[i] > bv) { bv = tmp[i]; best = i; }
      tmp[best] = -1e30f;
      ids[it] = best;
      wv[it]  = s_sc[best];
      wsum   += s_sc[best];
    }
    const float inv = 1.f / wsum;
    for (int it = 0; it < TOPK; ++it)
      combine[(size_t)t * NEXP + ids[it]] = wv[it] * inv;
  }
}

// ---------------- Kernel 2: deterministic per-expert token lists ----------------

__global__ void build_lists_kernel(const float* __restrict__ combine, int* __restrict__ counts,
                                   int* __restrict__ toks, float* __restrict__ wts) {
  const int e = blockIdx.x;
  int c = 0;
  for (int t = 0; t < T_TOK; ++t) {
    float v = combine[(size_t)t * NEXP + e];
    if (v != 0.f) {
      toks[e * T_TOK + c] = t;
      wts[e * T_TOK + c]  = v * ROUTED_SCALE;   // fold 2.5x routed scaling here
      ++c;
    }
  }
  counts[e] = c;
  for (int i = c; i < T_TOK; ++i) { toks[e * T_TOK + i] = 0; wts[e * T_TOK + i] = 0.f; }
}

// ---------------- Kernel 3: shared expert up-proj + relu^2 -> bf16 Hs ----------------
// grid (T/16, S/1024), block 256 (8 waves). Each wave: 8 accumulators over an N strip.

__global__ void shared_up_kernel(const float* __restrict__ x, const float* __restrict__ wsup,
                                 __bf16* __restrict__ Hs) {
  const int lane = threadIdx.x & 31;
  const int wave = threadIdx.x >> 5;
  const int m0 = blockIdx.x * 16;
  const int nblock = blockIdx.y * 1024;
  const float* xrow = x + (size_t)(m0 + (lane & 15)) * HDIM;

  v8f acc[8] = {};
  for (int k0 = 0; k0 < HDIM; k0 += 32) {
    if (k0 + 32 < HDIM)
      __builtin_prefetch(wsup + (size_t)(k0 + 32) * SDIM + nblock + wave * 128, 0, 1);
    v16bf a = load_a_f32row(xrow, k0, lane);
#pragma unroll
    for (int j = 0; j < 8; ++j) {
      const int n0 = nblock + (wave * 8 + j) * 16;
      v16bf b = load_b_f32(wsup, k0, n0, SDIM, lane);
      acc[j] = wmma_bf16(a, b, acc[j]);
    }
  }
#pragma unroll
  for (int j = 0; j < 8; ++j) {
    const int col = nblock + (wave * 8 + j) * 16 + (lane & 15);
#pragma unroll
    for (int i = 0; i < 8; ++i) {
      const int m = m0 + i + ((lane >> 4) << 3);
      float v = acc[j][i];
      v = v > 0.f ? v * v : 0.f;                        // relu^2
      Hs[(size_t)m * SDIM + col] = (__bf16)v;
    }
  }
}

// ---------------- Kernel 4: shared expert down-proj -> d_out (plain store, defines output) ----
// grid (T/16, H/256), block 256. Each wave covers 2 N-tiles, K = 8192.

__global__ void shared_down_kernel(const __bf16* __restrict__ Hs, const float* __restrict__ wsdown,
                                   float* __restrict__ out) {
  const int lane = threadIdx.x & 31;
  const int wave = threadIdx.x >> 5;
  const int m0 = blockIdx.x * 16;
  const int nbase = blockIdx.y * 256 + wave * 32;
  const __bf16* arow = Hs + (size_t)(m0 + (lane & 15)) * SDIM;

  v8f acc0 = {}, acc1 = {};
  for (int k0 = 0; k0 < SDIM; k0 += 32) {
    if (k0 + 32 < SDIM)
      __builtin_prefetch(wsdown + (size_t)(k0 + 32) * HDIM + nbase, 0, 1);
    v16bf a  = load_a_bf16row(arow, k0, lane);
    v16bf b0 = load_b_f32(wsdown, k0, nbase,      HDIM, lane);
    acc0 = wmma_bf16(a, b0, acc0);
    v16bf b1 = load_b_f32(wsdown, k0, nbase + 16, HDIM, lane);
    acc1 = wmma_bf16(a, b1, acc1);
  }
#pragma unroll
  for (int i = 0; i < 8; ++i) {
    const int m = m0 + i + ((lane >> 4) << 3);
    out[(size_t)m * HDIM + nbase +      (lane & 15)] = acc0[i];
    out[(size_t)m * HDIM + nbase + 16 + (lane & 15)] = acc1[i];
  }
}

// ---------------- Kernel 5: routed experts (grouped GEMM, M-tile = 16) ----------------
// grid (E, T/16), block 256 (8 waves). Up-GEMM into LDS (bf16, relu^2 * weight), then
// down-GEMM from LDS, atomic f32 accumulate into d_out.

__global__ void routed_kernel(const float* __restrict__ x, const float* __restrict__ w_up,
                              const float* __restrict__ w_down, const int* __restrict__ counts,
                              const int* __restrict__ toks, const float* __restrict__ wts,
                              float* __restrict__ out) {
  const int e  = blockIdx.x;
  const int mt = blockIdx.y;
  if (mt * 16 >= counts[e]) return;            // block-uniform early exit

  const int lane = threadIdx.x & 31;
  const int wave = threadIdx.x >> 5;

  __shared__ __bf16 Ys[16][IDIM];              // 32 KB: scaled relu^2 activations
  __shared__ int   stok[16];
  __shared__ float swt[16];
  if (threadIdx.x < 16) {
    stok[threadIdx.x] = toks[e * T_TOK + mt * 16 + threadIdx.x];
    swt[threadIdx.x]  = wts [e * T_TOK + mt * 16 + threadIdx.x];
  }
  __syncthreads();

  const float* We_up = w_up  + (size_t)e * HDIM * IDIM;   // [H x I]
  const float* We_dn = w_down + (size_t)e * IDIM * HDIM;  // [I x H]
  const float* xrow  = x + (size_t)stok[lane & 15] * HDIM;

  // ---- up: Y[16 x 1024] = wt * relu2(X_g @ Wup) ----
  {
    v8f acc[8] = {};
    for (int k0 = 0; k0 < HDIM; k0 += 32) {
      if (k0 + 32 < HDIM)
        __builtin_prefetch(We_up + (size_t)(k0 + 32) * IDIM + wave * 128, 0, 1);
      v16bf a = load_a_f32row(xrow, k0, lane);
#pragma unroll
      for (int j = 0; j < 8; ++j) {
        const int n0 = (wave * 8 + j) * 16;
        v16bf b = load_b_f32(We_up, k0, n0, IDIM, lane);
        acc[j] = wmma_bf16(a, b, acc[j]);
      }
    }
#pragma unroll
    for (int j = 0; j < 8; ++j) {
      const int col = (wave * 8 + j) * 16 + (lane & 15);
#pragma unroll
      for (int i = 0; i < 8; ++i) {
        const int m = i + ((lane >> 4) << 3);
        float v = acc[j][i];
        v = v > 0.f ? v * v : 0.f;
        Ys[m][col] = (__bf16)(v * swt[m]);     // padded rows have wt=0 -> zero
      }
    }
  }
  __syncthreads();

  // ---- down: out[tok] += Y @ Wdown  (two 1024-column halves) ----
  const __bf16* yrow = &Ys[lane & 15][0];
  for (int half = 0; half < 2; ++half) {
    v8f acc[8] = {};
    for (int k0 = 0; k0 < IDIM; k0 += 32) {
      if (k0 + 32 < IDIM)
        __builtin_prefetch(We_dn + (size_t)(k0 + 32) * HDIM + half * 1024 + wave * 128, 0, 1);
      v16bf a = load_a_bf16row(yrow, k0, lane);
#pragma unroll
      for (int j = 0; j < 8; ++j) {
        const int n0 = half * 1024 + (wave * 8 + j) * 16;
        v16bf b = load_b_f32(We_dn, k0, n0, HDIM, lane);
        acc[j] = wmma_bf16(a, b, acc[j]);
      }
    }
#pragma unroll
    for (int j = 0; j < 8; ++j) {
      const int col = half * 1024 + (wave * 8 + j) * 16 + (lane & 15);
#pragma unroll
      for (int i = 0; i < 8; ++i) {
        const int m = i + ((lane >> 4) << 3);
        const int tk = stok[m];
        unsafeAtomicAdd(&out[(size_t)tk * HDIM + col], acc[j][i]);  // global_atomic_add_f32
      }
    }
  }
}

// ---------------- Host launcher ----------------

extern "C" void kernel_launch(void* const* d_in, const int* in_sizes, int n_in,
                              void* d_out, int out_size, void* d_ws, size_t ws_size,
                              hipStream_t stream) {
  (void)in_sizes; (void)n_in; (void)out_size; (void)ws_size;
  const float* x       = (const float*)d_in[0];
  const float* gate_w  = (const float*)d_in[1];
  const float* bias    = (const float*)d_in[2];
  const float* w_up    = (const float*)d_in[3];
  const float* w_down  = (const float*)d_in[4];
  const float* ws_up   = (const float*)d_in[5];
  const float* ws_down = (const float*)d_in[6];
  float* out = (float*)d_out;

  char* p = (char*)d_ws;
  float* combine = (float*)p;            p += (size_t)T_TOK * NEXP * sizeof(float);
  int*   counts  = (int*)p;              p += (size_t)NEXP * sizeof(int);
  p = (char*)(((uintptr_t)p + 255) & ~(uintptr_t)255);
  int*   toks    = (int*)p;              p += (size_t)NEXP * T_TOK * sizeof(int);
  float* wts     = (float*)p;            p += (size_t)NEXP * T_TOK * sizeof(float);
  p = (char*)(((uintptr_t)p + 255) & ~(uintptr_t)255);
  __bf16* Hs     = (__bf16*)p;           // T * S bf16 = 4 MB

  router_kernel<<<T_TOK, 32, 0, stream>>>(x, gate_w, bias, combine);
  build_lists_kernel<<<NEXP, 1, 0, stream>>>(combine, counts, toks, wts);
  shared_up_kernel<<<dim3(T_TOK / 16, SDIM / 1024), 256, 0, stream>>>(x, ws_up, Hs);
  shared_down_kernel<<<dim3(T_TOK / 16, HDIM / 256), 256, 0, stream>>>(Hs, ws_down, out);
  routed_kernel<<<dim3(NEXP, T_TOK / 16), 256, 0, stream>>>(x, w_up, w_down, counts, toks, wts, out);
}
